// CortexODE_35098472743487
// MI455X (gfx1250) — matrix operand order repaired and checked by reading, not compiled, not run on hardware
//
#include <hip/hip_runtime.h>

#define M_PTS   150000
#define DVOL    192
#define SLOPE   0.2f

typedef _Float16 v16h __attribute__((ext_vector_type(16)));
typedef _Float16 v8h  __attribute__((ext_vector_type(8)));
typedef float    v8f  __attribute__((ext_vector_type(8)));

// ---------------------------------------------------------------------------
// 2x2x2 average pool: src (2D)^3 -> dst D^3
// ---------------------------------------------------------------------------
__global__ void pool2x_kernel(const float* __restrict__ src, float* __restrict__ dst, int Dd) {
    int idx = blockIdx.x * blockDim.x + threadIdx.x;
    int total = Dd * Dd * Dd;
    if (idx >= total) return;
    int x = idx % Dd;
    int y = (idx / Dd) % Dd;
    int z = idx / (Dd * Dd);
    int S = Dd * 2;
    float s = 0.f;
#pragma unroll
    for (int dz = 0; dz < 2; ++dz)
#pragma unroll
        for (int dy = 0; dy < 2; ++dy)
#pragma unroll
            for (int dx = 0; dx < 2; ++dx)
                s += src[((size_t)(2 * z + dz) * S + (2 * y + dy)) * S + (2 * x + dx)];
    dst[idx] = s * 0.125f;
}

// ---------------------------------------------------------------------------
// Weight convert: src is (N x Ksrc) row-major f32 (out_features x in_features);
// dst is (Kpad x N) row-major f16 (B-matrix layout for D = A*B), zero-padded K.
// ---------------------------------------------------------------------------
__global__ void cvt_weight_kernel(const float* __restrict__ src, _Float16* __restrict__ dst,
                                  int N, int Ksrc, int Kpad) {
    int idx = blockIdx.x * blockDim.x + threadIdx.x;
    int total = N * Kpad;
    if (idx >= total) return;
    int k = idx / N;
    int n = idx % N;
    float v = (k < Ksrc) ? src[(size_t)n * Ksrc + k] : 0.f;
    dst[(size_t)k * N + n] = (_Float16)v;
}

// ---------------------------------------------------------------------------
// WMMA helpers (layouts per CDNA5 ISA 7.12.2, wave32)
// ---------------------------------------------------------------------------
__device__ inline v16h load_a_frag(const _Float16* base, int stride, int kt, int lane) {
    int m = lane & 15;
    int koff = (lane >> 4) << 3;                 // lanes 16-31 hold K+8
    const _Float16* p0 = base + m * stride + kt * 32 + koff;
    v8h lo = *(const v8h*)(p0);
    v8h hi = *(const v8h*)(p0 + 16);
    v16h r;
#pragma unroll
    for (int i = 0; i < 8; ++i) { r[i] = lo[i]; r[i + 8] = hi[i]; }
    return r;
}

__device__ inline v16h load_b_frag(const _Float16* B, int strideN, int kt, int ncol, int lane) {
    int n = ncol + (lane & 15);
    int k0 = kt * 32 + ((lane >> 4) << 3);       // lanes 16-31 hold K+8; elems 8-15 at K+16
    v16h r;
#pragma unroll
    for (int i = 0; i < 8; ++i) {
        r[i] = B[(size_t)(k0 + i) * strideN + n];
        r[i + 8] = B[(size_t)(k0 + 16 + i) * strideN + n];
    }
    return r;
}

template <int KT>
__device__ inline v8f wave_gemm(const _Float16* Alds, int strideA,
                                const _Float16* Bglb, int strideB,
                                int ncol, int lane) {
    v8f acc = {};
#pragma unroll
    for (int kt = 0; kt < KT; ++kt) {
        v16h a = load_a_frag(Alds, strideA, kt, lane);
        v16h b = load_b_frag(Bglb, strideB, kt, ncol, lane);
        acc = __builtin_amdgcn_wmma_f32_16x16x32_f16(false, a, false, b,
                                                     (short)0, acc, false, false);
    }
    return acc;
}

__device__ inline float leaky(float v) { return v >= 0.f ? v : SLOPE * v; }

__device__ inline float tri_sample(const float* __restrict__ vol, int Dq,
                                   float cx, float cy, float cz) {
    float sm = 0.5f * (float)(Dq - 1);
    float dm = (float)(Dq - 1);
    float fx = fminf(fmaxf((cx + 1.f) * sm, 0.f), dm);
    float fy = fminf(fmaxf((cy + 1.f) * sm, 0.f), dm);
    float fz = fminf(fmaxf((cz + 1.f) * sm, 0.f), dm);
    float x0f = floorf(fx), y0f = floorf(fy), z0f = floorf(fz);
    float wx = fx - x0f, wy = fy - y0f, wz = fz - z0f;
    int x0 = (int)x0f, y0 = (int)y0f, z0 = (int)z0f;
    int x1 = min(x0 + 1, Dq - 1), y1 = min(y0 + 1, Dq - 1), z1 = min(z0 + 1, Dq - 1);
    size_t D = (size_t)Dq;
    const float* pz0 = vol + (size_t)z0 * D * D;
    const float* pz1 = vol + (size_t)z1 * D * D;
    float v000 = pz0[(size_t)y0 * D + x0], v001 = pz0[(size_t)y0 * D + x1];
    float v010 = pz0[(size_t)y1 * D + x0], v011 = pz0[(size_t)y1 * D + x1];
    float v100 = pz1[(size_t)y0 * D + x0], v101 = pz1[(size_t)y0 * D + x1];
    float v110 = pz1[(size_t)y1 * D + x0], v111 = pz1[(size_t)y1 * D + x1];
    float lo = (v000 * (1.f - wx) + v001 * wx) * (1.f - wy)
             + (v010 * (1.f - wx) + v011 * wx) * wy;
    float hi = (v100 * (1.f - wx) + v101 * wx) * (1.f - wy)
             + (v110 * (1.f - wx) + v111 * wx) * wy;
    return lo * (1.f - wz) + hi * wz;
}

// ---------------------------------------------------------------------------
// Mega kernel: one block = 16 points. sample -> conv -> localfc -> [fc1 concat]
// -> fc2 -> fc3 -> fc4.  256 threads = 8 waves.
// ---------------------------------------------------------------------------
__global__ __launch_bounds__(256)
void cortexode_fused_kernel(const float* __restrict__ x,
                            const float* __restrict__ V0,
                            const float* __restrict__ V1,
                            const float* __restrict__ V2,
                            const _Float16* __restrict__ convW,   // 384 x 128
                            const _Float16* __restrict__ lfW,     // 128 x 128
                            const _Float16* __restrict__ fc2W,    // 256 x 512
                            const _Float16* __restrict__ fc3W,    // 512 x 256
                            const float* __restrict__ fc1_w,      // 128 x 3
                            const float* __restrict__ fc1_b,
                            const float* __restrict__ conv_b,
                            const float* __restrict__ lf_b,
                            const float* __restrict__ fc2_b,
                            const float* __restrict__ fc3_b,
                            const float* __restrict__ fc4_w,      // 3 x 256
                            const float* __restrict__ fc4_b,
                            float* __restrict__ out) {
    __shared__ float xs[16][3];
    __shared__ __align__(32) _Float16 Acube[16][384];  // sampled features (K padded 375->384)
    __shared__ __align__(32) _Float16 Zl[16][128];     // conv output
    __shared__ __align__(32) _Float16 Zcat[16][256];   // [fc1 | localfc]
    __shared__ __align__(32) _Float16 Z2[16][512];     // fc2 output
    __shared__ __align__(32) _Float16 Z3[16][256];     // fc3 output

    const int t = threadIdx.x;
    const int lane = t & 31;
    const int wave = t >> 5;
    const int m0 = blockIdx.x * 16;

    // ---- stage 0: load the 16 points' coordinates --------------------------
    if (t < 48) xs[t / 3][t % 3] = x[(size_t)(m0 + t / 3) * 3 + (t % 3)];
    __syncthreads();

    // ---- stage 1: z_point = leaky(x @ fc1_w.T + fc1_b) -> Zcat[:, 0:128] ---
#pragma unroll
    for (int u = 0; u < 8; ++u) {
        int idx = u * 256 + t;          // 2048 = 16 x 128
        int p = idx >> 7;
        int n = idx & 127;
        float v = fc1_b[n] + xs[p][0] * fc1_w[n * 3 + 0]
                           + xs[p][1] * fc1_w[n * 3 + 1]
                           + xs[p][2] * fc1_w[n * 3 + 2];
        Zcat[p][n] = (_Float16)leaky(v);
    }

    // ---- stage 2: cube sampling -> Acube (16 x 384, f16) -------------------
    for (int it = 0; it < 24; ++it) {
        int sid = it * 256 + t;         // 6144 = 16 x 384
        int p = sid / 384;
        int kp = sid % 384;
        if (kp >= 375) { Acube[p][kp] = (_Float16)0.f; continue; }
        int q = kp / 125;
        int s = kp % 125;
        int i = s / 25, j = (s / 5) % 5, kk = s % 5;
        // g = linspace(-5//2, 5//2, 5) = linspace(-3, 2, 5) = -3 + 1.25*i
        float gi = -3.f + 1.25f * (float)i;
        float gj = -3.f + 1.25f * (float)j;
        float gk = -3.f + 1.25f * (float)kk;
        float scale = (float)(2 << q) / (float)DVOL;   // 2^(q+1)/192
        float cx = xs[p][0] + gi * scale;
        float cy = xs[p][1] + gj * scale;
        float cz = xs[p][2] + gk * scale;
        const float* vol = (q == 0) ? V0 : (q == 1) ? V1 : V2;
        int Dq = DVOL >> q;
        Acube[p][kp] = (_Float16)tri_sample(vol, Dq, cx, cy, cz);
    }
    __syncthreads();

    // ---- stage 3: conv einsum, (16x384) @ (384x128) + conv_b -> Zl ---------
    {
        int ncol = wave * 16;                       // 8 waves cover N = 128
        v8f acc = wave_gemm<12>(&Acube[0][0], 384, convW, 128, ncol, lane);
        int n = lane & 15;
        float bias = conv_b[ncol + n];
        int mb = (lane >> 4) * 8;
#pragma unroll
        for (int r = 0; r < 8; ++r)
            Zl[mb + r][ncol + n] = (_Float16)(acc[r] + bias);
    }
    __syncthreads();

    // ---- stage 4: localfc, (16x128) @ (128x128) + lf_b -> Zcat[:,128:256] --
    {
        int ncol = wave * 16;
        v8f acc = wave_gemm<4>(&Zl[0][0], 128, lfW, 128, ncol, lane);
        int n = lane & 15;
        float bias = lf_b[ncol + n];
        int mb = (lane >> 4) * 8;
#pragma unroll
        for (int r = 0; r < 8; ++r)
            Zcat[mb + r][128 + ncol + n] = (_Float16)(acc[r] + bias);
    }
    __syncthreads();

    // ---- stage 5: fc2, leaky((16x256) @ (256x512) + b) -> Z2 ---------------
#pragma unroll
    for (int rep = 0; rep < 4; ++rep) {
        int ncol = (rep * 8 + wave) * 16;           // 32 tiles over 8 waves
        v8f acc = wave_gemm<8>(&Zcat[0][0], 256, fc2W, 512, ncol, lane);
        int n = lane & 15;
        float bias = fc2_b[ncol + n];
        int mb = (lane >> 4) * 8;
#pragma unroll
        for (int r = 0; r < 8; ++r)
            Z2[mb + r][ncol + n] = (_Float16)leaky(acc[r] + bias);
    }
    __syncthreads();

    // ---- stage 6: fc3, leaky((16x512) @ (512x256) + b) -> Z3 ---------------
#pragma unroll
    for (int rep = 0; rep < 2; ++rep) {
        int ncol = (rep * 8 + wave) * 16;           // 16 tiles over 8 waves
        v8f acc = wave_gemm<16>(&Z2[0][0], 512, fc3W, 256, ncol, lane);
        int n = lane & 15;
        float bias = fc3_b[ncol + n];
        int mb = (lane >> 4) * 8;
#pragma unroll
        for (int r = 0; r < 8; ++r)
            Z3[mb + r][ncol + n] = (_Float16)leaky(acc[r] + bias);
    }
    __syncthreads();

    // ---- stage 7: fc4, (16x256) @ (256x3) + b -> out (f32) -----------------
    if (t < 48) {
        int p = t / 3;
        int col = t % 3;
        float acc = fc4_b[col];
        const float* w = fc4_w + (size_t)col * 256;
#pragma unroll 8
        for (int k = 0; k < 256; ++k)
            acc += (float)Z3[p][k] * w[k];
        out[(size_t)(m0 + p) * 3 + col] = acc;
    }
}

// ---------------------------------------------------------------------------
// Host side
// ---------------------------------------------------------------------------
extern "C" void kernel_launch(void* const* d_in, const int* in_sizes, int n_in,
                              void* d_out, int out_size, void* d_ws, size_t ws_size,
                              hipStream_t stream) {
    (void)in_sizes; (void)n_in; (void)out_size; (void)ws_size;
    const float* x      = (const float*)d_in[1];
    const float* V      = (const float*)d_in[2];
    const float* fc1_w  = (const float*)d_in[3];
    const float* fc1_b  = (const float*)d_in[4];
    const float* fc2_w  = (const float*)d_in[5];
    const float* fc2_b  = (const float*)d_in[6];
    const float* fc3_w  = (const float*)d_in[7];
    const float* fc3_b  = (const float*)d_in[8];
    const float* fc4_w  = (const float*)d_in[9];
    const float* fc4_b  = (const float*)d_in[10];
    const float* conv_w = (const float*)d_in[11];
    const float* conv_b = (const float*)d_in[12];
    const float* lf_w   = (const float*)d_in[13];
    const float* lf_b   = (const float*)d_in[14];
    float* out = (float*)d_out;

    char* ws = (char*)d_ws;
    float*     vol1  = (float*)(ws + 0);                 //  96^3 f32 = 3538944 B
    float*     vol2  = (float*)(ws + 3538944);           //  48^3 f32 =  442368 B
    _Float16*  convW = (_Float16*)(ws + 3981312);        // 384x128 f16
    _Float16*  lfW   = (_Float16*)(ws + 4079616);        // 128x128 f16
    _Float16*  fc2W  = (_Float16*)(ws + 4112384);        // 256x512 f16
    _Float16*  fc3W  = (_Float16*)(ws + 4374528);        // 512x256 f16

    // pooled pyramid
    pool2x_kernel<<<(96 * 96 * 96 + 255) / 256, 256, 0, stream>>>(V, vol1, 96);
    pool2x_kernel<<<(48 * 48 * 48 + 255) / 256, 256, 0, stream>>>(vol1, vol2, 48);

    // f16 weight staging (transposed to K x N, K padded for conv)
    cvt_weight_kernel<<<(384 * 128 + 255) / 256, 256, 0, stream>>>(conv_w, convW, 128, 375, 384);
    cvt_weight_kernel<<<(128 * 128 + 255) / 256, 256, 0, stream>>>(lf_w,   lfW,   128, 128, 128);
    cvt_weight_kernel<<<(256 * 512 + 255) / 256, 256, 0, stream>>>(fc2_w,  fc2W,  512, 256, 256);
    cvt_weight_kernel<<<(512 * 256 + 255) / 256, 256, 0, stream>>>(fc3_w,  fc3W,  256, 512, 512);

    // fully fused pipeline: 16 points per block, 8 waves
    cortexode_fused_kernel<<<M_PTS / 16, 256, 0, stream>>>(
        x, V, vol1, vol2, convW, lfW, fc2W, fc3W,
        fc1_w, fc1_b, conv_b, lf_b, fc2_b, fc3_b, fc4_w, fc4_b, out);
}
